// BertEncoder_89893665505785
// MI455X (gfx1250) — matrix operand order repaired and checked by reading, not compiled, not run on hardware
//
#include <hip/hip_runtime.h>
#include <cstdint>
#include <cstddef>

// ---------------- problem constants (BERT-base, B=16,S=512) ----------------
#define BDIM  16
#define SDIM  512
#define HDIM  768
#define LDIM  12
#define FDIM  3072
#define NHEAD 12
#define HDH   64            // head dim
#define MTOK  (BDIM * SDIM) // 8192 token rows

typedef __attribute__((ext_vector_type(16))) __bf16 v16bf;
typedef __attribute__((ext_vector_type(8)))  __bf16 v8bf;
typedef __attribute__((ext_vector_type(8)))  float  v8f;

static __device__ __forceinline__ __bf16 f2bf(float f) {
  unsigned u = __builtin_bit_cast(unsigned, f);
  unsigned r = u + 0x7FFFu + ((u >> 16) & 1u);   // round-to-nearest-even
  unsigned short h = (unsigned short)(r >> 16);
  return __builtin_bit_cast(__bf16, h);
}
static __device__ __forceinline__ v8f v8f_zero() {
  v8f z = {0.f, 0.f, 0.f, 0.f, 0.f, 0.f, 0.f, 0.f};
  return z;
}
static __device__ __forceinline__ v16bf cat8(v8bf lo, v8bf hi) {
  return __builtin_shufflevector(lo, hi, 0, 1, 2, 3, 4, 5, 6, 7,
                                 8, 9, 10, 11, 12, 13, 14, 15);
}

// ---------------- fp32 -> bf16 weight conversion ----------------
__global__ void k_f32_to_bf16(const float* __restrict__ in,
                              __bf16* __restrict__ out, long n) {
  long i = (long)blockIdx.x * blockDim.x + threadIdx.x;
  long stride = (long)gridDim.x * blockDim.x;
  for (; i < n; i += stride) out[i] = f2bf(in[i]);
}

// ---------------- embedding gather + LayerNorm (eps 1e-12) ----------------
__global__ void k_embed_ln(const int* __restrict__ ids,
                           const float* __restrict__ wemb,
                           const float* __restrict__ pemb,
                           const float* __restrict__ temb,
                           const float* __restrict__ g,
                           const float* __restrict__ bta,
                           float* __restrict__ x) {
  const int row = blockIdx.x;       // token row 0..MTOK-1
  const int t   = threadIdx.x;      // 256 threads, 3 cols each
  const int s   = row % SDIM;
  const int id  = ids[row];
  __shared__ float red[256];

  float v[3];
  float sum = 0.f;
#pragma unroll
  for (int j = 0; j < 3; ++j) {
    int c = t + j * 256;
    v[j] = wemb[(long)id * HDIM + c] + temb[c] + pemb[(long)s * HDIM + c];
    sum += v[j];
  }
  red[t] = sum; __syncthreads();
  for (int o = 128; o > 0; o >>= 1) { if (t < o) red[t] += red[t + o]; __syncthreads(); }
  const float mu = red[0] * (1.f / HDIM);
  __syncthreads();
  float ss = 0.f;
#pragma unroll
  for (int j = 0; j < 3; ++j) { float d = v[j] - mu; ss += d * d; }
  red[t] = ss; __syncthreads();
  for (int o = 128; o > 0; o >>= 1) { if (t < o) red[t] += red[t + o]; __syncthreads(); }
  const float rstd = rsqrtf(red[0] * (1.f / HDIM) + 1e-12f);
#pragma unroll
  for (int j = 0; j < 3; ++j) {
    int c = t + j * 256;
    x[(long)row * HDIM + c] = (v[j] - mu) * rstd * g[c] + bta[c];
  }
}

// ---------------- LayerNorm fp32 -> bf16 (eps 1e-5) ----------------
__global__ void k_ln_bf16(const float* __restrict__ x,
                          const float* __restrict__ g,
                          const float* __restrict__ bta,
                          __bf16* __restrict__ h) {
  const int row = blockIdx.x;
  const int t   = threadIdx.x;
  __shared__ float red[256];

  float v[3];
  float sum = 0.f;
#pragma unroll
  for (int j = 0; j < 3; ++j) {
    int c = t + j * 256;
    v[j] = x[(long)row * HDIM + c];
    sum += v[j];
  }
  red[t] = sum; __syncthreads();
  for (int o = 128; o > 0; o >>= 1) { if (t < o) red[t] += red[t + o]; __syncthreads(); }
  const float mu = red[0] * (1.f / HDIM);
  __syncthreads();
  float ss = 0.f;
#pragma unroll
  for (int j = 0; j < 3; ++j) { float d = v[j] - mu; ss += d * d; }
  red[t] = ss; __syncthreads();
  for (int o = 128; o > 0; o >>= 1) { if (t < o) red[t] += red[t + o]; __syncthreads(); }
  const float rstd = rsqrtf(red[0] * (1.f / HDIM) + 1e-5f);
#pragma unroll
  for (int j = 0; j < 3; ++j) {
    int c = t + j * 256;
    h[(long)row * HDIM + c] = f2bf((v[j] - mu) * rstd * g[c] + bta[c]);
  }
}

// ---------------- WMMA GEMM: C[M,N] = A[M,K](bf16) x W[K,N](bf16) ----------------
// 256 threads = 8 waves; block tile 128x128; each wave a 32x64 tile (8 WMMAs/k-step).
template <bool GELU, bool RES, bool OUTBF>
__global__ void k_gemm(const __bf16* __restrict__ A,
                       const __bf16* __restrict__ W,
                       const float*  __restrict__ bias,
                       const float*  __restrict__ res,
                       void* __restrict__ outp, int N, int K) {
  __shared__ uint4  sA4[128 * 4];      // 128 x 32 bf16 A tile (row-major)
  __shared__ __bf16 sBt[128 * 32];     // 128 x 32 bf16: W tile transposed (col-major)

  const int tid  = threadIdx.x;
  const int lane = tid & 31;
  const int wid  = tid >> 5;
  const int hi   = (lane >> 4) & 1;    // half-wave select
  const int n16  = lane & 15;
  const int m0   = blockIdx.y * 128;
  const int n0   = blockIdx.x * 128;
  const int wy   = wid & 3;            // 4 row groups of 32
  const int wx   = wid >> 2;           // 2 col groups of 64

  v8f c[2][4];
#pragma unroll
  for (int fr = 0; fr < 2; ++fr)
#pragma unroll
    for (int fc = 0; fc < 4; ++fc) c[fr][fc] = v8f_zero();

  const int arow = tid >> 1, ahalf = tid & 1;   // A staging: 16 bf16 / thread
  const int bk   = tid & 31, bc = tid >> 5;     // W staging: 16 bf16 / thread
  const __bf16* sA = (const __bf16*)sA4;
  const int kb8 = hi * 8;                       // A-frag K base (ISA 16-bit A layout)
  const int bb  = hi * 16;                      // unused: B frag uses kb8 (same k order as A)

  for (int kt = 0; kt < K; kt += 32) {
    // stage A tile (coalesced 16B loads)
    const uint4* ga = (const uint4*)(A + (long)(m0 + arow) * K + kt + ahalf * 16);
    sA4[arow * 4 + ahalf * 2 + 0] = ga[0];
    sA4[arow * 4 + ahalf * 2 + 1] = ga[1];
    // stage W tile transposed: sBt[col][k]; thread covers k-row bk, 16 cols
    {
      const uint4* gw = (const uint4*)(W + (long)(kt + bk) * N + n0 + bc * 16);
      uint4 w0 = gw[0], w1 = gw[1];
      const __bf16* e0 = (const __bf16*)&w0;
      const __bf16* e1 = (const __bf16*)&w1;
#pragma unroll
      for (int j = 0; j < 8; ++j) {
        sBt[(bc * 16 + j) * 32 + bk]     = e0[j];
        sBt[(bc * 16 + 8 + j) * 32 + bk] = e1[j];
      }
    }
    if (kt + 32 < K) {  // CDNA5 global_prefetch for next k-tile
      __builtin_prefetch(A + (long)(m0 + arow) * K + kt + 32, 0, 1);
      __builtin_prefetch(W + (long)(kt + 32 + bk) * N + n0, 0, 1);
    }
    __syncthreads();

    // A fragments: two contiguous 8-elem runs per lane (ds_load_b128)
    v16bf a0, a1;
    {
      const __bf16* r0 = sA + (wy * 32 + n16) * 32;
      const __bf16* r1 = sA + (wy * 32 + 16 + n16) * 32;
      a0 = cat8(*(const v8bf*)(r0 + kb8), *(const v8bf*)(r0 + 16 + kb8));
      a1 = cat8(*(const v8bf*)(r1 + kb8), *(const v8bf*)(r1 + 16 + kb8));
    }
#pragma unroll
    for (int fc = 0; fc < 4; ++fc) {
      // B fragment: two 8-elem k runs (8h.., 16+8h..) in transposed tile (2x ds_load_b128)
      v16bf bfr = cat8(*(const v8bf*)(sBt + (wx * 64 + fc * 16 + n16) * 32 + kb8), *(const v8bf*)(sBt + (wx * 64 + fc * 16 + n16) * 32 + 16 + kb8));
      c[0][fc] = __builtin_amdgcn_wmma_f32_16x16x32_bf16(false, a0, false, bfr, (short)0, c[0][fc], false, false);
      c[1][fc] = __builtin_amdgcn_wmma_f32_16x16x32_bf16(false, a1, false, bfr, (short)0, c[1][fc], false, false);
    }
    __syncthreads();
  }

  // epilogue: bias [+gelu] [+residual], store fp32 or bf16
  float* outF = (float*)outp;
  __bf16* outB = (__bf16*)outp;
#pragma unroll
  for (int fr = 0; fr < 2; ++fr) {
#pragma unroll
    for (int fc = 0; fc < 4; ++fc) {
      const int gcol = n0 + wx * 64 + fc * 16 + n16;
      const float bs = bias[gcol];
#pragma unroll
      for (int r = 0; r < 8; ++r) {
        const int grow = m0 + wy * 32 + fr * 16 + r + 8 * hi;
        float val = c[fr][fc][r] + bs;
        if (GELU) val = 0.5f * val * (1.f + erff(val * 0.70710678118654752f));
        if (RES)  val += res[(long)grow * N + gcol];
        if (OUTBF) outB[(long)grow * N + gcol] = f2bf(val);
        else       outF[(long)grow * N + gcol] = val;
      }
    }
  }
}

// ---------------- flash attention ----------------
// 128 threads = 4 waves; all 4 waves share (batch, head) and cover 4 adjacent
// 16-row q tiles, so K/V chunks are cooperatively staged in LDS once per block.
__global__ void k_attn(const __bf16* __restrict__ Q,
                       const __bf16* __restrict__ Kb,
                       const __bf16* __restrict__ V,
                       __bf16* __restrict__ O) {
  __shared__ __bf16 sK[32 * 64];     // [krow][d]      row-major
  __shared__ __bf16 sVt[64 * 32];    // [d][krow]      transposed
  __shared__ __bf16 Pl[4][16 * 32];  // per-wave P tile (C-layout -> A-layout)

  const int tid  = threadIdx.x;
  const int lane = tid & 31;
  const int wid  = tid >> 5;
  const int hi   = (lane >> 4) & 1;
  const int n16  = lane & 15;

  const int gw = blockIdx.x * 4 + wid;        // 0..6143
  const int qt = gw & 31;                     // q tile within sequence
  const int bh = gw >> 5;                     // shared by all 4 waves in block
  const int h  = bh % NHEAD;
  const int b  = bh / NHEAD;

  auto rowix = [&](int trow, int d) -> long {
    return ((long)(b * SDIM + trow)) * HDIM + h * HDH + d;
  };

  const int srow = tid >> 2, spart = tid & 3;  // staging: 32 rows x 4 parts of 16

  // Q fragments for head-dim chunks [0,32) and [32,64); unscaled (1/8 folded
  // into the f32 softmax below since positive scaling commutes with max/exp).
  v16bf qa[2];
  {
    const int m = qt * 16 + n16;
#pragma unroll
    for (int cch = 0; cch < 2; ++cch) {
      v8bf lo = *(const v8bf*)(Q + rowix(m, cch * 32 + hi * 8));
      v8bf hh = *(const v8bf*)(Q + rowix(m, cch * 32 + 16 + hi * 8));
      qa[cch] = cat8(lo, hh);
    }
  }

  float rmax[8], rsum[8];
  v8f acc[4];
#pragma unroll
  for (int r = 0; r < 8; ++r) { rmax[r] = -1e30f; rsum[r] = 0.f; }
#pragma unroll
  for (int f = 0; f < 4; ++f) acc[f] = v8f_zero();

  __bf16* P = Pl[wid];

  for (int kc = 0; kc < SDIM; kc += 32) {
    // cooperative staging of K (row-major) and V (transposed) chunks
    {
      const uint4* gk = (const uint4*)(Kb + rowix(kc + srow, spart * 16));
      ((uint4*)sK)[srow * 8 + spart * 2 + 0] = gk[0];
      ((uint4*)sK)[srow * 8 + spart * 2 + 1] = gk[1];
      const uint4* gv = (const uint4*)(V + rowix(kc + srow, spart * 16));
      uint4 v0 = gv[0], v1 = gv[1];
      const __bf16* e0 = (const __bf16*)&v0;
      const __bf16* e1 = (const __bf16*)&v1;
#pragma unroll
      for (int j = 0; j < 8; ++j) {
        sVt[(spart * 16 + j) * 32 + srow]     = e0[j];
        sVt[(spart * 16 + 8 + j) * 32 + srow] = e1[j];
      }
    }
    __syncthreads();

    // scores: two 16x16 tiles, K-dim=64 => 2 chained WMMAs each
    v8f s0 = v8f_zero(), s1 = v8f_zero();
#pragma unroll
    for (int t = 0; t < 2; ++t) {
      v8f sc = v8f_zero();
#pragma unroll
      for (int cch = 0; cch < 2; ++cch) {
        v16bf kf = cat8(*(const v8bf*)(sK + (t * 16 + n16) * 64 + cch * 32 + hi * 8), *(const v8bf*)(sK + (t * 16 + n16) * 64 + cch * 32 + 16 + hi * 8));
        sc = __builtin_amdgcn_wmma_f32_16x16x32_bf16(false, qa[cch], false, kf, (short)0, sc, false, false);
      }
      if (t == 0) s0 = sc; else s1 = sc;
    }

    // online softmax (rows r+8*hi live in the half-wave => width-16 reductions)
    float corr[8];
#pragma unroll
    for (int r = 0; r < 8; ++r) {
      float mx = fmaxf(s0[r], s1[r]);
      for (int o = 8; o; o >>= 1) mx = fmaxf(mx, __shfl_xor(mx, o, 16));
      const float mnew = fmaxf(rmax[r], mx * 0.125f);
      const float cr = __expf(rmax[r] - mnew);
      const float p0 = __expf(s0[r] * 0.125f - mnew);
      const float p1 = __expf(s1[r] * 0.125f - mnew);
      float ps = p0 + p1;
      for (int o = 8; o; o >>= 1) ps += __shfl_xor(ps, o, 16);
      rsum[r] = rsum[r] * cr + ps;
      rmax[r] = mnew;
      corr[r] = cr;
      P[(r + 8 * hi) * 32 + n16]      = f2bf(p0);
      P[(r + 8 * hi) * 32 + 16 + n16] = f2bf(p1);
    }
#pragma unroll
    for (int f = 0; f < 4; ++f)
#pragma unroll
      for (int r = 0; r < 8; ++r) acc[f][r] *= corr[r];

    // P (C-layout in LDS) -> A-fragment; DS ops are in-order per wave
    asm volatile("" ::: "memory");
    v16bf pa = cat8(*(const v8bf*)(P + n16 * 32 + hi * 8),
                    *(const v8bf*)(P + n16 * 32 + 16 + hi * 8));

    // O += P(16x32) x V(32x64): 4 WMMAs over head-dim columns
#pragma unroll
    for (int f = 0; f < 4; ++f) {
      v16bf vf = cat8(*(const v8bf*)(sVt + (f * 16 + n16) * 32 + hi * 8), *(const v8bf*)(sVt + (f * 16 + n16) * 32 + 16 + hi * 8));
      acc[f] = __builtin_amdgcn_wmma_f32_16x16x32_bf16(false, pa, false, vf, (short)0, acc[f], false, false);
    }
    __syncthreads();
  }

  // normalize and write bf16 attention output
#pragma unroll
  for (int f = 0; f < 4; ++f)
#pragma unroll
    for (int r = 0; r < 8; ++r) {
      const int m = qt * 16 + r + 8 * hi;
      O[rowix(m, f * 16 + n16)] = f2bf(acc[f][r] / rsum[r]);
    }
}

// ---------------- host orchestration ----------------
extern "C" void kernel_launch(void* const* d_in, const int* in_sizes, int n_in,
                              void* d_out, int out_size, void* d_ws, size_t ws_size,
                              hipStream_t stream) {
  (void)in_sizes; (void)n_in; (void)out_size; (void)ws_size;

  const int*   ids  = (const int*)  d_in[0];
  const float* wemb = (const float*)d_in[1];
  const float* pemb = (const float*)d_in[2];
  const float* temb = (const float*)d_in[3];
  const float* lnes = (const float*)d_in[4];
  const float* lneb = (const float*)d_in[5];
  const float* ln1s = (const float*)d_in[6];
  const float* ln1b = (const float*)d_in[7];
  const float* Wq   = (const float*)d_in[8];
  const float* bq   = (const float*)d_in[9];
  const float* Wk   = (const float*)d_in[10];
  const float* bk   = (const float*)d_in[11];
  const float* Wv   = (const float*)d_in[12];
  const float* bv   = (const float*)d_in[13];
  const float* Wo   = (const float*)d_in[14];
  const float* bo   = (const float*)d_in[15];
  const float* ln2s = (const float*)d_in[16];
  const float* ln2b = (const float*)d_in[17];
  const float* W1   = (const float*)d_in[18];
  const float* b1   = (const float*)d_in[19];
  const float* W2   = (const float*)d_in[20];
  const float* b2   = (const float*)d_in[21];

  char* ws = (char*)d_ws;
  auto take = [&](size_t bytes) -> char* {
    char* p = ws;
    ws += (bytes + 255) & ~(size_t)255;
    return p;
  };
  const size_t WHH = (size_t)LDIM * HDIM * HDIM;
  const size_t WHF = (size_t)LDIM * HDIM * FDIM;

  __bf16* wqb = (__bf16*)take(WHH * 2);
  __bf16* wkb = (__bf16*)take(WHH * 2);
  __bf16* wvb = (__bf16*)take(WHH * 2);
  __bf16* wob = (__bf16*)take(WHH * 2);
  __bf16* w1b = (__bf16*)take(WHF * 2);
  __bf16* w2b = (__bf16*)take(WHF * 2);
  float*  x   = (float*) take((size_t)MTOK * HDIM * 4);
  __bf16* hbf = (__bf16*)take((size_t)MTOK * HDIM * 2);
  __bf16* qb  = (__bf16*)take((size_t)MTOK * HDIM * 2);
  __bf16* kb  = (__bf16*)take((size_t)MTOK * HDIM * 2);
  __bf16* vb  = (__bf16*)take((size_t)MTOK * HDIM * 2);
  __bf16* ao  = (__bf16*)take((size_t)MTOK * HDIM * 2);
  __bf16* ff  = (__bf16*)take((size_t)MTOK * FDIM * 2);

  // weight conversion (deterministic each launch)
  k_f32_to_bf16<<<4096, 256, 0, stream>>>(Wq, wqb, (long)WHH);
  k_f32_to_bf16<<<4096, 256, 0, stream>>>(Wk, wkb, (long)WHH);
  k_f32_to_bf16<<<4096, 256, 0, stream>>>(Wv, wvb, (long)WHH);
  k_f32_to_bf16<<<4096, 256, 0, stream>>>(Wo, wob, (long)WHH);
  k_f32_to_bf16<<<4096, 256, 0, stream>>>(W1, w1b, (long)WHF);
  k_f32_to_bf16<<<4096, 256, 0, stream>>>(W2, w2b, (long)WHF);

  // embeddings + LN
  k_embed_ln<<<MTOK, 256, 0, stream>>>(ids, wemb, pemb, temb, lnes, lneb, x);

  const dim3 blk(256);
  const dim3 gH(HDIM / 128, MTOK / 128);
  const dim3 gF(FDIM / 128, MTOK / 128);

  for (int l = 0; l < LDIM; ++l) {
    const __bf16* wql = wqb + (size_t)l * HDIM * HDIM;
    const __bf16* wkl = wkb + (size_t)l * HDIM * HDIM;
    const __bf16* wvl = wvb + (size_t)l * HDIM * HDIM;
    const __bf16* wol = wob + (size_t)l * HDIM * HDIM;
    const __bf16* w1l = w1b + (size_t)l * HDIM * FDIM;
    const __bf16* w2l = w2b + (size_t)l * FDIM * HDIM;

    // h = LN1(x)
    k_ln_bf16<<<MTOK, 256, 0, stream>>>(x, ln1s + (size_t)l * HDIM, ln1b + (size_t)l * HDIM, hbf);
    // Q, K, V projections (bf16 out)
    k_gemm<false, false, true><<<gH, blk, 0, stream>>>(hbf, wql, bq + (size_t)l * HDIM, nullptr, qb, HDIM, HDIM);
    k_gemm<false, false, true><<<gH, blk, 0, stream>>>(hbf, wkl, bk + (size_t)l * HDIM, nullptr, kb, HDIM, HDIM);
    k_gemm<false, false, true><<<gH, blk, 0, stream>>>(hbf, wvl, bv + (size_t)l * HDIM, nullptr, vb, HDIM, HDIM);
    // flash attention
    k_attn<<<(BDIM * NHEAD * (SDIM / 16)) / 4, 128, 0, stream>>>(qb, kb, vb, ao);
    // x = x + attn @ Wo + bo  (fp32 out, fused residual)
    k_gemm<false, true, false><<<gH, blk, 0, stream>>>(ao, wol, bo + (size_t)l * HDIM, x, x, HDIM, HDIM);
    // h2 = LN2(x)
    k_ln_bf16<<<MTOK, 256, 0, stream>>>(x, ln2s + (size_t)l * HDIM, ln2b + (size_t)l * HDIM, hbf);
    // ff = gelu(h2 @ W1 + b1) (bf16 out)
    k_gemm<true, false, true><<<gF, blk, 0, stream>>>(hbf, w1l, b1 + (size_t)l * FDIM, nullptr, ff, FDIM, HDIM);
    // x = x + ff @ W2 + b2 (fp32 out; last layer writes d_out directly)
    float* outx = (l == LDIM - 1) ? (float*)d_out : x;
    k_gemm<false, true, false><<<gH, blk, 0, stream>>>(ff, w2l, b2 + (size_t)l * HDIM, x, outx, HDIM, FDIM);
  }
}
